// Linear_28879360098368
// MI455X (gfx1250) — compile-verified, matrix operand order checked by reading, not run on hardware
//
#include <hip/hip_runtime.h>

// ---------------------------------------------------------------------------
// Blockwise-dequant GEMM for MI455X (gfx1250, wave32, WMMA).
//   out[M,N] = x[M,K] * (weight[N,K] . scale[N/128, K/128])^T
//   M = 4*2048 = 8192, K = 4096, N = 12288
// bf16 WMMA (16x16x32) with f32 accumulation — matches the reference's
// gemm_impl=='bf16' quantized path. Dequant scale is uniform per 128x32
// B-tile, folded into the global->LDS staging pass.
// ---------------------------------------------------------------------------

typedef __attribute__((ext_vector_type(16))) __bf16 v16bf;
typedef __attribute__((ext_vector_type(8)))  float  v8f;

#define M_DIM 8192
#define N_DIM 12288
#define K_DIM 4096
#define QBLK  128          // dequant block size
#define TM    128          // workgroup tile M
#define TN    128          // workgroup tile N
#define TK    32           // K step (bf16 WMMA K)
#define KP    40           // padded LDS row length in bf16 (80 bytes)

union FragU { v16bf v; uint4 u[2]; };
union AccU  { v8f   v; float f[8]; };

__global__ __launch_bounds__(256, 2) void gemm_dequant_bf16_wmma(
    const float* __restrict__ x,       // [M_DIM, K_DIM]
    const float* __restrict__ w,       // [N_DIM, K_DIM]
    const float* __restrict__ scale,   // [N_DIM/QBLK, K_DIM/QBLK]
    float* __restrict__ out)           // [M_DIM, N_DIM]
{
    __shared__ __bf16 ldsA[2][TM * KP];
    __shared__ __bf16 ldsB[2][TN * KP];

    const int tid   = threadIdx.x;
    const int tileN = blockIdx.x * TN;
    const int tileM = blockIdx.y * TM;

    const int lane  = tid & 31;
    const int wave  = tid >> 5;
    const int waveM = (wave & 3) * 32;   // 4 waves tile M: 32 rows each
    const int waveN = (wave >> 2) * 64;  // 2 waves tile N: 64 cols each
    const int halfl = lane >> 4;         // 0: lanes 0-15, 1: lanes 16-31
    const int lrow  = lane & 15;

    // 8 accumulators: 2 (M) x 4 (N) 16x16 f32 tiles per wave
    AccU acc[2][4];
#pragma unroll
    for (int mi = 0; mi < 2; ++mi)
#pragma unroll
        for (int ni = 0; ni < 4; ++ni)
#pragma unroll
            for (int r = 0; r < 8; ++r) acc[mi][ni].f[r] = 0.0f;

    const int scaleRowBase = (tileN >> 7) * (K_DIM / QBLK);

    // Register staging for the global->LDS pipeline (16 floats of A + B each)
    float4 aReg[4], bReg[4];

    auto globalLoad = [&](int k0) {
        // One dequant scale covers the whole 128x32 B tile (k window of 32
        // starting at a multiple of 32 never crosses a 128 block boundary).
        const float s = scale[scaleRowBase + (k0 >> 7)];
#pragma unroll
        for (int j = 0; j < 4; ++j) {
            const int idx = tid + j * 256;      // 0..1023 float4 slots
            const int row = idx >> 3;           // 0..127
            const int col = (idx & 7) * 4;      // 0..28
            aReg[j] = *(const float4*)(x + (size_t)(tileM + row) * K_DIM + k0 + col);
            float4 b = *(const float4*)(w + (size_t)(tileN + row) * K_DIM + k0 + col);
            b.x *= s; b.y *= s; b.z *= s; b.w *= s;
            bReg[j] = b;
        }
    };

    auto ldsStore = [&](int buf) {
#pragma unroll
        for (int j = 0; j < 4; ++j) {
            const int idx = tid + j * 256;
            const int row = idx >> 3;
            const int col = (idx & 7) * 4;
            union { __bf16 h[4]; uint2 u; } ta, tb;
            ta.h[0] = (__bf16)aReg[j].x; ta.h[1] = (__bf16)aReg[j].y;
            ta.h[2] = (__bf16)aReg[j].z; ta.h[3] = (__bf16)aReg[j].w;
            tb.h[0] = (__bf16)bReg[j].x; tb.h[1] = (__bf16)bReg[j].y;
            tb.h[2] = (__bf16)bReg[j].z; tb.h[3] = (__bf16)bReg[j].w;
            *(uint2*)&ldsA[buf][row * KP + col] = ta.u;   // 8B aligned (KP*2=80)
            *(uint2*)&ldsB[buf][row * KP + col] = tb.u;
        }
    };

    auto compute = [&](int buf) {
        FragU a[2], b[4];
        // ISA 16-bit A-fragment layout: lane (halfl,lrow) holds row M=lrow,
        // K = halfl*8 + {0..7} in VGPR0-3 and K = 16 + halfl*8 + {0..7} in
        // VGPR4-7 -> two contiguous 16B LDS loads per fragment. B tile is
        // stored [N][K] so the same gather yields the B fragment.
#pragma unroll
        for (int mi = 0; mi < 2; ++mi) {
            const __bf16* p = &ldsA[buf][(waveM + mi * 16 + lrow) * KP + halfl * 8];
            a[mi].u[0] = *(const uint4*)(p);
            a[mi].u[1] = *(const uint4*)(p + 16);
        }
#pragma unroll
        for (int ni = 0; ni < 4; ++ni) {
            const __bf16* p = &ldsB[buf][(waveN + ni * 16 + lrow) * KP + halfl * 8];
            b[ni].u[0] = *(const uint4*)(p);
            b[ni].u[1] = *(const uint4*)(p + 16);
        }
#pragma unroll
        for (int mi = 0; mi < 2; ++mi)
#pragma unroll
            for (int ni = 0; ni < 4; ++ni)
                acc[mi][ni].v = __builtin_amdgcn_wmma_f32_16x16x32_bf16(
                    /*neg_a=*/false, a[mi].v,
                    /*neg_b=*/false, b[ni].v,
                    /*c_mod=*/(short)0, acc[mi][ni].v,
                    /*reuse_a=*/false, /*reuse_b=*/false);
    };

    // Software pipeline: global loads for step k+1 overlap WMMA on step k.
    globalLoad(0);
    ldsStore(0);
    __syncthreads();

    const int nk = K_DIM / TK;  // 128 iterations
    for (int kt = 1; kt < nk; ++kt) {
        globalLoad(kt * TK);          // prefetch next tile into registers
        compute((kt - 1) & 1);        // consume current LDS buffer
        ldsStore(kt & 1);             // fill the other buffer
        __syncthreads();
    }
    compute((nk - 1) & 1);

    // Store: C/D layout — VGPR r holds M = r + 8*halfl, N = lrow.
#pragma unroll
    for (int mi = 0; mi < 2; ++mi) {
#pragma unroll
        for (int ni = 0; ni < 4; ++ni) {
            const int col = tileN + waveN + ni * 16 + lrow;
#pragma unroll
            for (int r = 0; r < 8; ++r) {
                const int row = tileM + waveM + mi * 16 + r + halfl * 8;
                out[(size_t)row * N_DIM + col] = acc[mi][ni].f[r];
            }
        }
    }
}

extern "C" void kernel_launch(void* const* d_in, const int* in_sizes, int n_in,
                              void* d_out, int out_size, void* d_ws, size_t ws_size,
                              hipStream_t stream) {
    (void)in_sizes; (void)n_in; (void)out_size; (void)d_ws; (void)ws_size;
    const float* x     = (const float*)d_in[0];   // [4,2048,4096] -> [8192,4096]
    const float* w     = (const float*)d_in[1];   // [12288,4096]
    const float* scale = (const float*)d_in[2];   // [96,32]
    float*       out   = (float*)d_out;           // [8192,12288]

    dim3 grid(N_DIM / TN, M_DIM / TM);            // 96 x 64 workgroups
    gemm_dequant_bf16_wmma<<<grid, 256, 0, stream>>>(x, w, scale, out);
}
